// LSTMAutoencoder_6339371728927
// MI455X (gfx1250) — compile-verified, matrix operand order checked by reading, not run on hardware
//
#include <hip/hip_runtime.h>

typedef __attribute__((ext_vector_type(16))) _Float16 v16h;
typedef __attribute__((ext_vector_type(8)))  float    v8f;

#define T_LEN 256
#define D_IN  3
#define H_DIM 16

__device__ __forceinline__ float fast_sigmoid(float x) {
  float e = __builtin_amdgcn_exp2f(-1.44269504f * x);   // exp(-x)
  return __builtin_amdgcn_rcpf(1.0f + e);
}
__device__ __forceinline__ float fast_tanh(float x) {
  float e = __builtin_amdgcn_exp2f(2.88539008f * x);    // exp(2x)
  return 1.0f - 2.0f * __builtin_amdgcn_rcpf(e + 1.0f);
}

// v16h element e of the A/B fragment maps to K = e + 8*hi + (e>=8 ? 8 : 0)
// (16-bit 16x32 A layout; B 32x16 is identical with lane%16 = column N).
__device__ __forceinline__ v16h load_gate_frag(const float* __restrict__ Wih, int din,
                                               const float* __restrict__ Whh,
                                               int g, int n, int hi) {
  v16h a;
#pragma unroll
  for (int e = 0; e < 16; ++e) {
    int k = e + 8 * hi + ((e >= 8) ? 8 : 0);
    float v;
    if (k < 16) v = (k < din) ? Wih[(16 * g + n) * din + k] : 0.0f;  // x-part, zero pad
    else        v = Whh[(16 * g + n) * H_DIM + (k - 16)];            // h-part
    a[e] = (_Float16)v;
  }
  return a;
}

__global__ __launch_bounds__(128) void lstm_ae_fused(
    const float* __restrict__ x,
    const float* __restrict__ eWih, const float* __restrict__ eWhh, const float* __restrict__ eb,
    const float* __restrict__ dWih, const float* __restrict__ dWhh, const float* __restrict__ db,
    const float* __restrict__ oW,  const float* __restrict__ ob,
    float* __restrict__ out, int numTiles) {
  const int lane = threadIdx.x & 31;
  const int wave = threadIdx.x >> 5;
  const int tile = blockIdx.x * (blockDim.x >> 5) + wave;
  if (tile >= numTiles) return;          // wave-uniform: EXEC stays all-ones for WMMA
  const int n  = lane & 15;              // A-fragment row / B-fragment column index
  const int hi = lane >> 4;              // which K-half this lane carries
  const int b  = tile * 16 + n;          // batch row handled by this lane-column

  // ================= encoder =================
  v16h A[4];
  float bias[4][8];
#pragma unroll
  for (int g = 0; g < 4; ++g) {
    A[g] = load_gate_frag(eWih, D_IN, eWhh, g, n, hi);
#pragma unroll
    for (int r = 0; r < 8; ++r) bias[g][r] = eb[16 * g + r + 8 * hi];
  }

  v16h Bf = {};            // [x | h]^T fragment: elements 0..7 = x slice, 8..15 = h slice
  float h[8], c[8];
#pragma unroll
  for (int r = 0; r < 8; ++r) { h[r] = 0.f; c[r] = 0.f; }

  const float* xrow = x + (size_t)b * T_LEN * D_IN;
  float x0 = 0.f, x1 = 0.f, x2 = 0.f;
  if (hi == 0) { x0 = xrow[0]; x1 = xrow[1]; x2 = xrow[2]; }  // K=0..2; K>=3 stays 0

  for (int t = 0; t < T_LEN; ++t) {
    Bf[0] = (_Float16)x0; Bf[1] = (_Float16)x1; Bf[2] = (_Float16)x2;
    float nx0 = 0.f, nx1 = 0.f, nx2 = 0.f;                 // prefetch t+1
    if (hi == 0 && t + 1 < T_LEN) {
      const float* p = xrow + (t + 1) * D_IN;
      nx0 = p[0]; nx1 = p[1]; nx2 = p[2];
    }
    v8f acc[4];
#pragma unroll
    for (int g = 0; g < 4; ++g) {
#pragma unroll
      for (int r = 0; r < 8; ++r) acc[g][r] = bias[g][r];  // C preloaded with bias
      acc[g] = __builtin_amdgcn_wmma_f32_16x16x32_f16(false, A[g], false, Bf,
                                                      (short)0, acc[g], false, false);
    }
#pragma unroll
    for (int r = 0; r < 8; ++r) {                          // D layout == next B h-slice
      float iv = fast_sigmoid(acc[0][r]);
      float fv = fast_sigmoid(acc[1][r]);
      float gv = fast_tanh(acc[2][r]);
      float ov = fast_sigmoid(acc[3][r]);
      c[r] = fv * c[r] + iv * gv;
      h[r] = ov * fast_tanh(c[r]);
      Bf[8 + r] = (_Float16)h[r];
    }
    x0 = nx0; x1 = nx1; x2 = nx2;
  }

  // ================= decoder + linear head =================
#pragma unroll
  for (int g = 0; g < 4; ++g) {
    A[g] = load_gate_frag(dWih, H_DIM, dWhh, g, n, hi);
#pragma unroll
    for (int r = 0; r < 8; ++r) bias[g][r] = db[16 * g + r + 8 * hi];
  }
  v16h Ao;                                  // head: rows d<3 = out_W over K=16..31
#pragma unroll
  for (int e = 0; e < 16; ++e) {
    int k = e + 8 * hi + ((e >= 8) ? 8 : 0);
    float v = 0.f;
    if (k >= 16 && n < D_IN) v = oW[n * H_DIM + (k - 16)];
    Ao[e] = (_Float16)v;
  }
  const float ob0 = ob[0], ob1 = ob[1], ob2 = ob[2];

#pragma unroll
  for (int r = 0; r < 8; ++r) {
    Bf[r]     = (_Float16)h[r];   // latent (encoder final h) -> constant K=0..15 slice
    Bf[8 + r] = (_Float16)0.f;    // decoder h starts at 0
    h[r] = 0.f; c[r] = 0.f;
  }

  float* orow = out + (size_t)b * T_LEN * D_IN;
  for (int t = 0; t < T_LEN; ++t) {
    v8f acc[4];
#pragma unroll
    for (int g = 0; g < 4; ++g) {
#pragma unroll
      for (int r = 0; r < 8; ++r) acc[g][r] = bias[g][r];
      acc[g] = __builtin_amdgcn_wmma_f32_16x16x32_f16(false, A[g], false, Bf,
                                                      (short)0, acc[g], false, false);
    }
#pragma unroll
    for (int r = 0; r < 8; ++r) {
      float iv = fast_sigmoid(acc[0][r]);
      float fv = fast_sigmoid(acc[1][r]);
      float gv = fast_tanh(acc[2][r]);
      float ov = fast_sigmoid(acc[3][r]);
      c[r] = fv * c[r] + iv * gv;
      h[r] = ov * fast_tanh(c[r]);
      Bf[8 + r] = (_Float16)h[r];    // h_t feeds both the head below and step t+1
    }
    v8f accO = {};
    accO = __builtin_amdgcn_wmma_f32_16x16x32_f16(false, Ao, false, Bf,
                                                  (short)0, accO, false, false);
    if (hi == 0) {                    // lane holds out[d=r][b] for r=0..2
      float* p = orow + t * D_IN;
      p[0] = accO[0] + ob0;
      p[1] = accO[1] + ob1;
      p[2] = accO[2] + ob2;
    }
  }
}

extern "C" void kernel_launch(void* const* d_in, const int* in_sizes, int n_in,
                              void* d_out, int out_size, void* d_ws, size_t ws_size,
                              hipStream_t stream) {
  const float* x    = (const float*)d_in[0];
  const float* eWih = (const float*)d_in[1];
  const float* eWhh = (const float*)d_in[2];
  const float* eb   = (const float*)d_in[3];
  const float* dWih = (const float*)d_in[4];
  const float* dWhh = (const float*)d_in[5];
  const float* db   = (const float*)d_in[6];
  const float* oW   = (const float*)d_in[7];
  const float* ob   = (const float*)d_in[8];
  float* out = (float*)d_out;

  int B = in_sizes[0] / (T_LEN * D_IN);
  int numTiles = B / 16;                 // B = 8192 -> 512 waves
  const int wavesPerBlock = 4;           // 128 threads = 4 wave32
  int blocks = (numTiles + wavesPerBlock - 1) / wavesPerBlock;
  lstm_ae_fused<<<blocks, wavesPerBlock * 32, 0, stream>>>(
      x, eWih, eWhh, eb, dWih, dWhh, db, oW, ob, out, numTiles);
}
